// RPN_83966610637103
// MI455X (gfx1250) — matrix operand order repaired and checked by reading, not compiled, not run on hardware
//
#include <hip/hip_runtime.h>
#include <hip/hip_bf16.h>
#include <stdint.h>

typedef __attribute__((ext_vector_type(16))) __bf16 v16bf;
typedef __attribute__((ext_vector_type(8)))  float  v8f;

#define NEGV (-1000000000.0f)

__device__ __forceinline__ unsigned short f2bf(float f) {
  unsigned u = __float_as_uint(f);
  unsigned r = u + 0x7FFFu + ((u >> 16) & 1u);   // round-to-nearest-even
  return (unsigned short)(r >> 16);
}

union Frag16 { uint4 q[2]; unsigned u[8]; v16bf v; };

// ---------------------------------------------------------------------------
// Weight prep: conv_w (OIHW fp32) -> bf16 [pos][oc][ic]  (A-fragment friendly)
// ---------------------------------------------------------------------------
__global__ void prep_w_kernel(const float* __restrict__ cw,
                              unsigned short* __restrict__ wbf) {
  int i = blockIdx.x * blockDim.x + threadIdx.x;
  if (i >= 9 * 256 * 256) return;
  int p  = i >> 16;
  int oc = (i >> 8) & 255;
  int ic = i & 255;
  int ky = p / 3, kx = p % 3;
  wbf[i] = f2bf(cw[((oc * 256 + ic) * 3 + ky) * 3 + kx]);
}

// ---------------------------------------------------------------------------
// Feature prep: x (NCHW fp32) -> bf16 NHWC [pix][ic]  (B-fragment friendly;
// whole tensor stays L2-resident: all levels total ~45 MB bf16 << 192 MB L2)
// ---------------------------------------------------------------------------
__global__ void prep_x_kernel(const float* __restrict__ x, int HW,
                              unsigned short* __restrict__ xh) {
  int t = blockIdx.x * blockDim.x + threadIdx.x;
  if (t >= HW * 32) return;
  int pix = t >> 5;
  int icg = (t & 31) * 8;
  const float* src = x + (size_t)icg * HW + pix;
  unsigned u0, u1, u2, u3;
  u0 = (unsigned)f2bf(src[0])              | ((unsigned)f2bf(src[(size_t)1 * HW]) << 16);
  u1 = (unsigned)f2bf(src[(size_t)2 * HW]) | ((unsigned)f2bf(src[(size_t)3 * HW]) << 16);
  u2 = (unsigned)f2bf(src[(size_t)4 * HW]) | ((unsigned)f2bf(src[(size_t)5 * HW]) << 16);
  u3 = (unsigned)f2bf(src[(size_t)6 * HW]) | ((unsigned)f2bf(src[(size_t)7 * HW]) << 16);
  *(uint4*)(xh + (((size_t)pix) << 8) + icg) = make_uint4(u0, u1, u2, u3);
}

// ---------------------------------------------------------------------------
// Fused conv3x3 (bf16 WMMA, f32 accum) + ReLU + 1x1 heads (obj + delta)
// One block = 32-pixel row segment x all 256 output channels.
// A fragments: contiguous b128 global loads from prearranged bf16 weights.
// B fragments: contiguous b128 global loads from bf16 NHWC (EXEC-masked halo).
// Head weights: async global->LDS copy hidden behind the WMMA loop.
// ---------------------------------------------------------------------------
__global__ __launch_bounds__(256) void conv_head_kernel(
    const unsigned short* __restrict__ xh, int H, int W, int tilesX,
    const unsigned short* __restrict__ wbf,
    const float* __restrict__ conv_b,
    const float* __restrict__ obj_w,   const float* __restrict__ obj_b,
    const float* __restrict__ delta_w, const float* __restrict__ delta_b,
    float* __restrict__ scores, float* __restrict__ deltas)
{
  __shared__ __attribute__((aligned(16))) float HWs[15 * 256]; // 3 obj + 12 delta
  __shared__ float hred[15 * 32];                              // cross-wave head acc

  const int tid  = threadIdx.x;
  const int wave = tid >> 5, lane = tid & 31;
  const int y  = blockIdx.x / tilesX;
  const int x0 = (blockIdx.x % tilesX) * 32;

  // Kick off async LDS fill of the head-weight table (pure byte copy, f32).
  // obj_w: 3*256 f32 = 192 x 16B slots; delta_w: 12*256 f32 = 768 x 16B slots.
  if (tid < 192) {
    unsigned ldsoff = (unsigned)(uintptr_t)((char*)HWs + tid * 16);
    unsigned long long ga = (unsigned long long)(uintptr_t)((const char*)obj_w + tid * 16);
    asm volatile("global_load_async_to_lds_b128 %0, %1, off"
                 :: "v"(ldsoff), "v"(ga) : "memory");
  }
  for (int s = tid; s < 768; s += 256) {
    unsigned ldsoff = (unsigned)(uintptr_t)((char*)HWs + 3072 + s * 16);
    unsigned long long ga = (unsigned long long)(uintptr_t)((const char*)delta_w + s * 16);
    asm volatile("global_load_async_to_lds_b128 %0, %1, off"
                 :: "v"(ldsoff), "v"(ga) : "memory");
  }
  for (int i = tid; i < 15 * 32; i += 256) {
    int ch = i >> 5;
    hred[i] = (ch < 3) ? obj_b[ch] : delta_b[ch - 3];
  }

  v8f acc00 = {}, acc01 = {}, acc10 = {}, acc11 = {};
  const int m   = lane & 15;
  const int khi = lane >> 4;
  const int ocb = wave * 32;

  for (int chunk = 0; chunk < 8; ++chunk) {
    const int ic0 = chunk * 32;
    #pragma unroll
    for (int p = 0; p < 9; ++p) {
      const int ky = p / 3, kx = p % 3;
      Frag16 a0, a1, b0, b1;
      // A: weights, 2x 16B per fragment per lane (L2-resident)
      size_t abase0 = ((size_t)(p * 256 + ocb + m) << 8) + (size_t)(ic0 + khi * 16);
      a0.q[0] = *(const uint4*)(wbf + abase0);
      a0.q[1] = *(const uint4*)(wbf + abase0 + 8);
      size_t abase1 = abase0 + (16u << 8);
      a1.q[0] = *(const uint4*)(wbf + abase1);
      a1.q[1] = *(const uint4*)(wbf + abase1 + 8);

      // B: NHWC bf16, 2x 16B per fragment per lane, EXEC-masked zero halo
      const int gy = y + ky - 1;
      const bool rowok = (gy >= 0) && (gy < H);
      const int gx0 = x0 + (lane & 15) + kx - 1;
      #pragma unroll
      for (int j = 0; j < 8; ++j) { b0.u[j] = 0u; b1.u[j] = 0u; }
      if (rowok && gx0 >= 0 && gx0 < W) {
        size_t base = (((size_t)gy * W + gx0) << 8) + (size_t)(ic0 + khi * 16);
        b0.q[0] = *(const uint4*)(xh + base);
        b0.q[1] = *(const uint4*)(xh + base + 8);
      }
      const int gx1 = gx0 + 16;
      if (rowok && gx1 < W) {
        size_t base = (((size_t)gy * W + gx1) << 8) + (size_t)(ic0 + khi * 16);
        b1.q[0] = *(const uint4*)(xh + base);
        b1.q[1] = *(const uint4*)(xh + base + 8);
      }

      acc00 = __builtin_amdgcn_wmma_f32_16x16x32_bf16(false, a0.v, false, b0.v, (short)0, acc00, false, false);
      acc01 = __builtin_amdgcn_wmma_f32_16x16x32_bf16(false, a0.v, false, b1.v, (short)0, acc01, false, false);
      acc10 = __builtin_amdgcn_wmma_f32_16x16x32_bf16(false, a1.v, false, b0.v, (short)0, acc10, false, false);
      acc11 = __builtin_amdgcn_wmma_f32_16x16x32_bf16(false, a1.v, false, b1.v, (short)0, acc11, false, false);
    }
  }

  // Async head-weight copy must have landed; then workgroup-sync.
  asm volatile("s_wait_asynccnt 0x0" ::: "memory");
  __syncthreads();

  // bias + relu + head partial sums (per lane: 16 output channels, 2 pixels)
  float ps[15][2];
  #pragma unroll
  for (int ch = 0; ch < 15; ++ch) { ps[ch][0] = 0.f; ps[ch][1] = 0.f; }
  #pragma unroll
  for (int oh = 0; oh < 2; ++oh) {
    #pragma unroll
    for (int r = 0; r < 8; ++r) {
      int oc = ocb + oh * 16 + khi * 8 + r;
      float bias = conv_b[oc];
      float h0 = fmaxf((oh ? acc10[r] : acc00[r]) + bias, 0.f);
      float h1 = fmaxf((oh ? acc11[r] : acc01[r]) + bias, 0.f);
      #pragma unroll
      for (int ch = 0; ch < 15; ++ch) {
        float w = HWs[ch * 256 + oc];
        ps[ch][0] += w * h0;
        ps[ch][1] += w * h1;
      }
    }
  }
  {
    int n = lane & 15;
    #pragma unroll
    for (int ch = 0; ch < 15; ++ch) {
      atomicAdd(&hred[ch * 32 + n],      ps[ch][0]);
      atomicAdd(&hred[ch * 32 + 16 + n], ps[ch][1]);
    }
  }
  __syncthreads();

  for (int i = tid; i < 15 * 32; i += 256) {
    int ch = i >> 5, pp = i & 31;
    int gx = x0 + pp;
    if (gx < W) {
      size_t pix = (size_t)y * W + gx;
      if (ch < 3) {
        scores[pix * 3 + ch] = hred[i];
      } else {
        int cc = ch - 3, a = cc >> 2, c = cc & 3;
        deltas[(pix * 3 + a) * 4 + c] = hred[i];
      }
    }
  }
}

// ---------------------------------------------------------------------------
// Exact top-K per level: 4-pass radix select on monotonic uint keys
// ---------------------------------------------------------------------------
__global__ __launch_bounds__(1024) void topk_kernel(
    const float* __restrict__ scores, int N, int K, int* __restrict__ cand_idx)
{
  __shared__ unsigned bins[256];
  __shared__ unsigned sPref, sPrefMask, cntG, cntE;
  __shared__ int sRem;
  const int tid = threadIdx.x;
  if (tid == 0) { sPref = 0; sPrefMask = 0; sRem = K; cntG = 0; cntE = 0; }
  __syncthreads();

  for (int pass = 0; pass < 4; ++pass) {
    int shift = 24 - 8 * pass;
    if (tid < 256) bins[tid] = 0;
    __syncthreads();
    unsigned pref = sPref, mask = sPrefMask;
    for (int i = tid; i < N; i += 1024) {
      unsigned u = __float_as_uint(scores[i]);
      unsigned k = (u & 0x80000000u) ? ~u : (u | 0x80000000u);
      if ((k & mask) == pref) atomicAdd(&bins[(k >> shift) & 255u], 1u);
    }
    __syncthreads();
    if (tid == 0) {
      int rem = sRem; unsigned cum = 0; int b;
      for (b = 255; b > 0; --b) {
        unsigned c = bins[b];
        if (cum + c >= (unsigned)rem) break;
        cum += c;
      }
      sRem = rem - (int)cum;
      sPref = pref | ((unsigned)b << shift);
      sPrefMask = mask | (0xFFu << shift);
    }
    __syncthreads();
  }
  unsigned thr = sPref;
  int rem = sRem;
  int kg  = K - rem;
  for (int i = tid; i < N; i += 1024) {
    unsigned u = __float_as_uint(scores[i]);
    unsigned k = (u & 0x80000000u) ? ~u : (u | 0x80000000u);
    if (k > thr) {
      unsigned p = atomicAdd(&cntG, 1u);
      cand_idx[p] = i;
    } else if (k == thr) {
      unsigned e = atomicAdd(&cntE, 1u);
      if ((int)e < rem) cand_idx[kg + (int)e] = i;
    }
  }
}

// ---------------------------------------------------------------------------
// Decode: analytic anchors + delta decode + clip + validity masking
// ---------------------------------------------------------------------------
__global__ void decode_kernel(const int* __restrict__ cand_idx,
    const float* __restrict__ scoresAll, const float* __restrict__ deltasAll,
    const int* __restrict__ ih, const int* __restrict__ iw,
    float* __restrict__ cand_score, float* __restrict__ cand_boxes)
{
  int t = blockIdx.x * blockDim.x + threadIdx.x;
  if (t >= 4768) return;
  const int   Wl[5]      = {256, 128, 64, 32, 16};
  const float strideL[5] = {4.f, 8.f, 16.f, 32.f, 64.f};
  const float scaleL[5]  = {32.f, 64.f, 128.f, 256.f, 512.f};
  const int scoreOff[5]  = {0, 196608, 245760, 258048, 261120};
  const int Nl[5]        = {196608, 49152, 12288, 3072, 768};

  int lvl = t / 1000; if (lvl > 4) lvl = 4;
  int idx = cand_idx[t];
  if (idx < 0) idx = 0;
  if (idx >= Nl[lvl]) idx = Nl[lvl] - 1;

  float score = scoresAll[scoreOff[lvl] + idx];
  const float* dl = deltasAll + (size_t)scoreOff[lvl] * 4 + (size_t)idx * 4;
  float dx = dl[0], dy = dl[1], dw = dl[2], dh = dl[3];

  int a = idx % 3, pix = idx / 3;
  int Wv = Wl[lvl];
  int xq = pix % Wv, yq = pix / Wv;
  const float sr[3] = {0.70710678118654752f, 1.0f, 1.41421356237309515f};
  float wsz = scaleL[lvl] / sr[a];
  float hsz = scaleL[lvl] * sr[a];
  float cxa = (xq + 0.5f) * strideL[lvl];
  float cya = (yq + 0.5f) * strideL[lvl];
  float cx = dx * wsz + cxa;
  float cy = dy * hsz + cya;
  float w  = expf(dw) * wsz;
  float h  = expf(dh) * hsz;
  float imw = (float)(*iw), imh = (float)(*ih);
  float x1 = fminf(fmaxf(cx - 0.5f * w, 0.f), imw);
  float y1 = fminf(fmaxf(cy - 0.5f * h, 0.f), imh);
  float x2 = fminf(fmaxf(cx + 0.5f * w, 0.f), imw);
  float y2 = fminf(fmaxf(cy + 0.5f * h, 0.f), imh);
  bool valid = (x2 > x1) && (y2 > y1);
  cand_score[t] = valid ? score : NEGV;
  cand_boxes[t * 4 + 0] = x1; cand_boxes[t * 4 + 1] = y1;
  cand_boxes[t * 4 + 2] = x2; cand_boxes[t * 4 + 3] = y2;
}

// ---------------------------------------------------------------------------
// Greedy NMS: 1 workgroup, scores/boxes register-resident, wave32 shfl argmax
// ---------------------------------------------------------------------------
__global__ __launch_bounds__(1024) void nms_kernel(
    const float* __restrict__ cand_score, const float* __restrict__ cand_boxes,
    float* __restrict__ out)
{
  const int N = 4768, T = 1024, S = 5;
  const int t = threadIdx.x;
  __shared__ float wv[32];
  __shared__ int   wi[32];
  __shared__ float bestVs;
  __shared__ int   bestIs;

  float ls[S], bx1[S], by1[S], bx2[S], by2[S], ar[S];
  #pragma unroll
  for (int s = 0; s < S; ++s) {
    int gi = t + s * T;
    if (gi < N) {
      ls[s]  = cand_score[gi];
      bx1[s] = cand_boxes[gi * 4 + 0]; by1[s] = cand_boxes[gi * 4 + 1];
      bx2[s] = cand_boxes[gi * 4 + 2]; by2[s] = cand_boxes[gi * 4 + 3];
      ar[s]  = fmaxf(bx2[s] - bx1[s], 0.f) * fmaxf(by2[s] - by1[s], 0.f);
    } else {
      ls[s] = NEGV; bx1[s] = by1[s] = bx2[s] = by2[s] = 0.f; ar[s] = 0.f;
    }
  }
  const int lane = t & 31, wave = t >> 5;

  for (int j = 0; j < 1000; ++j) {
    float v = ls[0]; int gi = t;
    #pragma unroll
    for (int s = 1; s < S; ++s)
      if (ls[s] > v) { v = ls[s]; gi = t + s * T; }
    #pragma unroll
    for (int off = 16; off > 0; off >>= 1) {
      float ov = __shfl_down(v, off, 32);
      int   oi = __shfl_down(gi, off, 32);
      if (ov > v || (ov == v && oi < gi)) { v = ov; gi = oi; }
    }
    if (lane == 0) { wv[wave] = v; wi[wave] = gi; }
    __syncthreads();
    if (wave == 0) {
      float v2 = wv[lane]; int i2 = wi[lane];
      #pragma unroll
      for (int off = 16; off > 0; off >>= 1) {
        float ov = __shfl_down(v2, off, 32);
        int   oi = __shfl_down(i2, off, 32);
        if (ov > v2 || (ov == v2 && oi < i2)) { v2 = ov; i2 = oi; }
      }
      if (lane == 0) { bestVs = v2; bestIs = i2; }
    }
    __syncthreads();
    float bv = bestVs; int bi = bestIs;
    float Bx1 = cand_boxes[bi * 4 + 0], By1 = cand_boxes[bi * 4 + 1];
    float Bx2 = cand_boxes[bi * 4 + 2], By2 = cand_boxes[bi * 4 + 3];
    float Bar = fmaxf(Bx2 - Bx1, 0.f) * fmaxf(By2 - By1, 0.f);
    bool valid = bv > NEGV * 0.5f;
    #pragma unroll
    for (int s = 0; s < S; ++s) {
      float ix1 = fmaxf(Bx1, bx1[s]);
      float iy1 = fmaxf(By1, by1[s]);
      float ix2 = fminf(Bx2, bx2[s]);
      float iy2 = fminf(By2, by2[s]);
      float inter = fmaxf(ix2 - ix1, 0.f) * fmaxf(iy2 - iy1, 0.f);
      float iou = inter / (ar[s] + Bar - inter + 1e-9f);
      if (iou > 0.7f) ls[s] = NEGV;
      if (t + s * T == bi) ls[s] = NEGV;
    }
    if (t == 0) {
      float mk = valid ? 1.f : 0.f;
      out[j * 5 + 0] = Bx1 * mk; out[j * 5 + 1] = By1 * mk;
      out[j * 5 + 2] = Bx2 * mk; out[j * 5 + 3] = By2 * mk;
      out[j * 5 + 4] = bv * mk;
    }
  }
}

// ---------------------------------------------------------------------------
extern "C" void kernel_launch(void* const* d_in, const int* in_sizes, int n_in,
                              void* d_out, int out_size, void* d_ws, size_t ws_size,
                              hipStream_t stream) {
  (void)in_sizes; (void)n_in; (void)out_size; (void)ws_size;
  const float* xs[5] = {(const float*)d_in[0], (const float*)d_in[1],
                        (const float*)d_in[2], (const float*)d_in[3],
                        (const float*)d_in[4]};
  const float* conv_w  = (const float*)d_in[5];
  const float* conv_b  = (const float*)d_in[6];
  const float* obj_w   = (const float*)d_in[7];
  const float* obj_b   = (const float*)d_in[8];
  const float* delta_w = (const float*)d_in[9];
  const float* delta_b = (const float*)d_in[10];
  const int*   ih      = (const int*)d_in[11];
  const int*   iw      = (const int*)d_in[12];

  char* ws = (char*)d_ws;
  unsigned short* wbf   = (unsigned short*)ws;                     // 1,179,648 B
  unsigned short* xhAll = (unsigned short*)(ws + 1179648);         // 44,695,552 B
  float* scoresAll = (float*)(ws + 1179648 + 44695552);            // 261,888 f
  float* deltasAll = scoresAll + 261888;                           // 1,047,552 f
  float* candScore = deltasAll + 1047552;                          // 4,768 f
  int*   candIdx   = (int*)(candScore + 4768);                     // 4,768 i
  float* candBoxes = (float*)(candIdx + 4768);                     // 19,072 f

  prep_w_kernel<<<(9 * 256 * 256 + 255) / 256, 256, 0, stream>>>(conv_w, wbf);

  const int Hs[5]       = {256, 128, 64, 32, 16};
  const int pixOff[5]   = {0, 65536, 81920, 86016, 87040};   // NHWC pixel offsets
  const int scoreOff[5] = {0, 196608, 245760, 258048, 261120};
  const int Ks[5]       = {1000, 1000, 1000, 1000, 768};
  const int candOff[5]  = {0, 1000, 2000, 3000, 4000};

  for (int l = 0; l < 5; ++l) {
    int HW = Hs[l] * Hs[l];
    prep_x_kernel<<<(HW * 32 + 255) / 256, 256, 0, stream>>>(
        xs[l], HW, xhAll + (size_t)pixOff[l] * 256);
  }
  for (int l = 0; l < 5; ++l) {
    int H = Hs[l], W = Hs[l];
    int tilesX = (W + 31) / 32;
    conv_head_kernel<<<H * tilesX, 256, 0, stream>>>(
        xhAll + (size_t)pixOff[l] * 256, H, W, tilesX, wbf,
        conv_b, obj_w, obj_b, delta_w, delta_b,
        scoresAll + scoreOff[l], deltasAll + (size_t)scoreOff[l] * 4);
  }
  for (int l = 0; l < 5; ++l) {
    int N = Hs[l] * Hs[l] * 3;
    topk_kernel<<<1, 1024, 0, stream>>>(scoresAll + scoreOff[l], N, Ks[l],
                                        candIdx + candOff[l]);
  }
  decode_kernel<<<(4768 + 255) / 256, 256, 0, stream>>>(
      candIdx, scoresAll, deltasAll, ih, iw, candScore, candBoxes);
  nms_kernel<<<1, 1024, 0, stream>>>(candScore, candBoxes, (float*)d_out);
}
